// SelfAttention_26414048870847
// MI455X (gfx1250) — matrix operand order repaired and checked by reading, not compiled, not run on hardware
//
#include <hip/hip_runtime.h>

// ---------------------------------------------------------------------------
// Self-attention: conv1d projections (3-tap GEMMs), q^T k, column softmax,
// v @ beta — all heavy math on v_wmma_f32_16x16x32_bf16 (gfx1250, wave32).
// ---------------------------------------------------------------------------

typedef __attribute__((ext_vector_type(16))) __bf16 v16bf;
typedef __attribute__((ext_vector_type(8)))  float  v8f;

__device__ __forceinline__ unsigned short f32_to_bf16(float f) {
  union { float f; unsigned u; } x; x.f = f;
  unsigned r = x.u + 0x7FFFu + ((x.u >> 16) & 1u);   // round-to-nearest-even
  return (unsigned short)(r >> 16);
}

// ---------------- pack / convert kernels -----------------------------------

// n must be a multiple of 4 (it is: B*C*L).
__global__ void cvt_f32_bf16_kernel(const float* __restrict__ src,
                                    unsigned int* __restrict__ dst, long n4) {
  long i = (long)blockIdx.x * blockDim.x + threadIdx.x;
  long stride = (long)gridDim.x * blockDim.x;
  for (; i < n4; i += stride) {
    float4 f = ((const float4*)src)[i];
    unsigned lo = (unsigned)f32_to_bf16(f.x) | ((unsigned)f32_to_bf16(f.y) << 16);
    unsigned hi = (unsigned)f32_to_bf16(f.z) | ((unsigned)f32_to_bf16(f.w) << 16);
    dst[2 * i]     = lo;
    dst[2 * i + 1] = hi;
  }
}

// w: [Cout][Cin][3] f32  ->  wp: [3][Cout][Cin] bf16
__global__ void pack_w_kernel(const float* __restrict__ w,
                              unsigned short* __restrict__ wp,
                              int Cout, int Cin) {
  long n = (long)Cout * Cin * 3;
  long stride = (long)gridDim.x * blockDim.x;
  for (long i = (long)blockIdx.x * blockDim.x + threadIdx.x; i < n; i += stride) {
    int  t  = (int)(i % 3);
    long oi = i / 3;                      // oi = o*Cin + ci
    wp[(long)t * Cout * Cin + oi] = f32_to_bf16(w[oi * 3 + t]);
  }
}

// ---------------- WMMA GEMM (optionally 3-tap conv) ------------------------
// O[m,n] = g * sum_t sum_k A_t[m,k] * B[k, n + t - HALO]
//   A: [NTAPS][M][Kt] bf16 row-major (aBatch = 0 for shared weights)
//   B: [Kt][N]        bf16 row-major
// OUTMODE: 0 -> f32 O[m,n] ; 1 -> bf16 O[m,n] ; 2 -> bf16 O[n,m] (transposed)
template<int NTAPS, int OUTMODE>
__global__ __launch_bounds__(256)
void gemm_wmma_kernel(const unsigned short* __restrict__ A,
                      const unsigned short* __restrict__ Bmat,
                      float* __restrict__ Cf, unsigned short* __restrict__ Cb,
                      int M, int N, int Kt,
                      long aBatch, long bBatch, long oBatch,
                      const float* __restrict__ gammap) {
  constexpr int BM = 64, BN = 64, BK = 64;
  constexpr int KP = 72;                       // padded LDS row, elems
  constexpr int HALO = (NTAPS > 1) ? 1 : 0;
  constexpr int HN = BN + 2 * HALO;

  __shared__ alignas(16) unsigned short As[NTAPS][BM][KP];
  __shared__ alignas(16) unsigned short Bt[HN][KP];   // B tile, transposed [n][k]

  const int bz = blockIdx.z;
  const int n0 = blockIdx.x * BN;
  const int m0 = blockIdx.y * BM;
  const unsigned short* Ab = A    + (long)bz * aBatch;
  const unsigned short* Bb = Bmat + (long)bz * bBatch;

  const int tid  = threadIdx.x;
  const int lane = tid & 31;
  const int wave = tid >> 5;
  const int wm   = wave >> 1;        // 0..3  (M subtile)
  const int wn   = wave & 1;         // 0..1  (N half, 32 cols each)
  const int lm   = lane & 15;
  const int half = lane >> 4;

  v8f acc0 = {}; v8f acc1 = {};
  union Frag { v16bf v; uint4 u[2]; };

  for (int k0 = 0; k0 < Kt; k0 += BK) {
    // ---- stage A tiles (per tap): 64 rows x 64 bf16 = 512 x uint4 ---------
#pragma unroll
    for (int j = 0; j < 2; ++j) {
      const int idx = tid + j * 256;
      const int row = idx >> 3, c8 = idx & 7;        // 8 x uint4 per row
#pragma unroll
      for (int t = 0; t < NTAPS; ++t) {
        const unsigned short* src =
            Ab + (long)t * M * Kt + (long)(m0 + row) * Kt + k0 + c8 * 8;
        *(uint4*)&As[t][row][c8 * 8] = *(const uint4*)src;
      }
    }
    // ---- stage B interior, vectorized: 64 k-rows x 64 n, transpose to LDS -
#pragma unroll
    for (int j = 0; j < 2; ++j) {
      const int idx = tid + j * 256;
      const int k = idx >> 3, c8 = idx & 7;          // 8 n-cols per uint4
      const uint4 d = *(const uint4*)&Bb[(long)(k0 + k) * N + n0 + c8 * 8];
      const unsigned short* e = (const unsigned short*)&d;
#pragma unroll
      for (int q = 0; q < 8; ++q) Bt[HALO + c8 * 8 + q][k] = e[q];
    }
    // ---- halo columns (conv only): n0-1 and n0+BN, zero-filled at edges ---
    if (NTAPS > 1 && tid < 2 * BK) {
      const int k = tid & (BK - 1), side = tid >> 6;  // BK == 64
      const int ng = side ? (n0 + BN) : (n0 - 1);
      unsigned short val = 0;
      if (ng >= 0 && ng < N) val = Bb[(long)(k0 + k) * N + ng];
      Bt[side ? (HN - 1) : 0][k] = val;
    }
    if (k0 + BK < Kt) {                              // global_prefetch_b8 hints
      __builtin_prefetch(Bb + (long)(k0 + BK) * N + n0 + tid, 0, 1);
      __builtin_prefetch(Ab + (long)m0 * Kt + k0 + BK + tid, 0, 1);
    }
    __syncthreads();

    // ---- two K-chunks of 32 per staged panel ------------------------------
#pragma unroll
    for (int c = 0; c < 2; ++c) {
      // Fetch ALL fragments of this chunk before the WMMA chain so the
      // scheduler can overlap ds returns with matrix ops (partial dscnt
      // waits instead of s_wait_dscnt 0 before every WMMA).
      // A fragment, 16-bit 16x32 layout: lane m=l%16;
      // K runs [half*8,+8) and [16+half*8,+8) -> two b128 LDS reads
      Frag a[NTAPS];
#pragma unroll
      for (int t = 0; t < NTAPS; ++t) {
        a[t].u[0] = *(const uint4*)&As[t][wm * 16 + lm][c * 32 + half * 8];
        a[t].u[1] = *(const uint4*)&As[t][wm * 16 + lm][c * 32 + 16 + half * 8];
      }
      // B fragment, 32x16 layout: lane n=l%16, K runs [half*8,+8) and [16+half*8,+8)
      Frag bfr[2][NTAPS];
#pragma unroll
      for (int s = 0; s < 2; ++s) {
#pragma unroll
        for (int t = 0; t < NTAPS; ++t) {
          const int h = wn * 32 + s * 16 + lm + (NTAPS > 1 ? t : 0);
          bfr[s][t].u[0] = *(const uint4*)&Bt[h][c * 32 + half * 8];
          bfr[s][t].u[1] = *(const uint4*)&Bt[h][c * 32 + 16 + half * 8];
        }
      }
#pragma unroll
      for (int s = 0; s < 2; ++s) {
#pragma unroll
        for (int t = 0; t < NTAPS; ++t) {
          v8f& acc = s ? acc1 : acc0;
          acc = __builtin_amdgcn_wmma_f32_16x16x32_bf16(
              false, a[t].v, false, bfr[s][t].v, (short)0, acc, false, false);
        }
      }
    }
    __syncthreads();
  }

  // ---- store: C/D layout: VGPR r, lanes 0-15 -> M=r, lanes 16-31 -> M=r+8
  const float g = gammap ? gammap[0] : 1.0f;
#pragma unroll
  for (int s = 0; s < 2; ++s) {
    const v8f& acc = s ? acc1 : acc0;
    const int n = n0 + wn * 32 + s * 16 + lm;
#pragma unroll
    for (int r = 0; r < 8; ++r) {
      const int m = m0 + wm * 16 + r + half * 8;
      const float val = acc[r] * g;
      if (OUTMODE == 0) Cf[(long)bz * oBatch + (long)m * N + n] = val;
      else if (OUTMODE == 1) Cb[(long)bz * oBatch + (long)m * N + n] = f32_to_bf16(val);
      else Cb[(long)bz * oBatch + (long)n * M + m] = f32_to_bf16(val);  // O^T
    }
  }
}

// ---------------- column softmax (softmax over i of scores[b,i,j]) ---------
__global__ __launch_bounds__(256)
void softmax_col_kernel(float* __restrict__ scores,
                        unsigned short* __restrict__ beta, int L) {
  const int b = blockIdx.y;
  const int j = blockIdx.x * blockDim.x + threadIdx.x;
  float* s = scores + (long)b * L * L;
  unsigned short* bt = beta + (long)b * L * L;
  float mx = -3.402823466e38f;
  for (int i = 0; i < L; ++i) mx = fmaxf(mx, s[(long)i * L + j]);
  float sum = 0.f;
  for (int i = 0; i < L; ++i) {
    float e = __expf(s[(long)i * L + j] - mx);
    s[(long)i * L + j] = e;                        // in-place scratch
    sum += e;
  }
  const float inv = 1.0f / sum;
  for (int i = 0; i < L; ++i)
    bt[(long)i * L + j] = f32_to_bf16(s[(long)i * L + j] * inv);
}

// ---------------------------------------------------------------------------

extern "C" void kernel_launch(void* const* d_in, const int* in_sizes, int n_in,
                              void* d_out, int out_size, void* d_ws, size_t ws_size,
                              hipStream_t stream) {
  (void)in_sizes; (void)n_in; (void)out_size; (void)ws_size;
  const int B = 8, C = 512, L = 2048, CQ = 64;

  const float* x     = (const float*)d_in[0];
  const float* Wq    = (const float*)d_in[1];
  const float* Wk    = (const float*)d_in[2];
  const float* Wv    = (const float*)d_in[3];
  const float* gamma = (const float*)d_in[4];
  float* out = (float*)d_out;

  char* ws = (char*)d_ws;
  auto alloc = [&](size_t bytes) -> char* {
    char* p = ws; ws += (bytes + 255) & ~(size_t)255; return p;
  };
  unsigned short* xbf    = (unsigned short*)alloc((size_t)B * C * L * 2);
  unsigned short* wqp    = (unsigned short*)alloc((size_t)3 * CQ * C * 2);
  unsigned short* wkp    = (unsigned short*)alloc((size_t)3 * CQ * C * 2);
  unsigned short* wvp    = (unsigned short*)alloc((size_t)3 * C * C * 2);
  unsigned short* qT     = (unsigned short*)alloc((size_t)B * L * CQ * 2);  // [L][CQ]
  unsigned short* kbf    = (unsigned short*)alloc((size_t)B * CQ * L * 2);  // [CQ][L]
  unsigned short* vbf    = (unsigned short*)alloc((size_t)B * C * L * 2);   // [C][L]
  float*          scores = (float*)         alloc((size_t)B * L * L * 4);
  unsigned short* beta   = (unsigned short*)alloc((size_t)B * L * L * 2);

  // 1) packing
  cvt_f32_bf16_kernel<<<1024, 256, 0, stream>>>(x, (unsigned int*)xbf,
                                                (long)B * C * L / 4);
  pack_w_kernel<<<256, 256, 0, stream>>>(Wq, wqp, CQ, C);
  pack_w_kernel<<<256, 256, 0, stream>>>(Wk, wkp, CQ, C);
  pack_w_kernel<<<512, 256, 0, stream>>>(Wv, wvp, C,  C);

  // 2) convolutions as 3-tap WMMA GEMMs (A = W taps, B = x, K = Cin)
  gemm_wmma_kernel<3, 2><<<dim3(L / 64, CQ / 64, B), 256, 0, stream>>>(
      wqp, xbf, nullptr, qT, CQ, L, C, 0, (long)C * L, (long)L * CQ, nullptr);
  gemm_wmma_kernel<3, 1><<<dim3(L / 64, CQ / 64, B), 256, 0, stream>>>(
      wkp, xbf, nullptr, kbf, CQ, L, C, 0, (long)C * L, (long)CQ * L, nullptr);
  gemm_wmma_kernel<3, 1><<<dim3(L / 64, C / 64, B), 256, 0, stream>>>(
      wvp, xbf, nullptr, vbf, C, L, C, 0, (long)C * L, (long)C * L, nullptr);

  // 3) scores = q^T k : A = qT [L][CQ], B = k [CQ][L], K = CQ
  gemm_wmma_kernel<1, 0><<<dim3(L / 64, L / 64, B), 256, 0, stream>>>(
      qT, kbf, scores, nullptr, L, L, CQ,
      (long)L * CQ, (long)CQ * L, (long)L * L, nullptr);

  // 4) softmax over axis=1 (column-wise over i)
  softmax_col_kernel<<<dim3(L / 256, B), 256, 0, stream>>>(scores, beta, L);

  // 5) out = gamma * v @ beta : A = v [C][L], B = beta [L][L], K = L
  gemm_wmma_kernel<1, 0><<<dim3(L / 64, C / 64, B), 256, 0, stream>>>(
      vbf, beta, out, nullptr, C, L, L,
      (long)C * L, (long)L * L, (long)C * L, gamma);
}
